// MonarchMixerSequenceMixingFlashFFTConv_52767968199450
// MI455X (gfx1250) — matrix operand order repaired and checked by reading, not compiled, not run on hardware
//
#include <hip/hip_runtime.h>

// ---------------------------------------------------------------------------
// Monarch Mixer sequence mixing (FlashFFTConv math) for MI455X / gfx1250.
//   B=4, L=2048, D=768, 3D=2304, fp32 in/out.
//
//  k_split x3  : fp32 -> bf16 hi/lo planes for u, in_w, out_w (mem-bound)
//  k_inproj    : x[b,o,l] = sum_d in_w[o,d]*u[b,l,d]   bf16x3 WMMA, 64x32/wave
//  k_shortconv : depthwise conv3+bias; x1v = x1*v, x2  elementwise
//  k_longconv  : causal Toeplitz conv per (b,d) + gate f32 WMMA 16x16x4,
//                LDS operands staged via TDM tensor_load_to_lds
//  k_outproj   : out[b,l,o] = sum_c y2[b,l,c]*out_w[o,c] + out_b[o]
// ---------------------------------------------------------------------------

#define B_   4
#define L_   2048
#define D_   768
#define D3_  2304

typedef unsigned short u16;
typedef __attribute__((ext_vector_type(16))) __bf16 v16bf;
typedef __attribute__((ext_vector_type(8)))  float  v8f;
typedef __attribute__((ext_vector_type(2)))  float  v2f;
typedef __attribute__((ext_vector_type(4)))  unsigned int v4u;
typedef __attribute__((ext_vector_type(8)))  int    v8i;
typedef __attribute__((ext_vector_type(4)))  int    v4i;

// ---- TDM availability (this toolchain: 6-arg clang-23 builtin) ------------
#ifndef USE_TDM
#define USE_TDM 1
#endif
#if defined(__has_builtin)
# if USE_TDM && __has_builtin(__builtin_amdgcn_tensor_load_to_lds)
#  define TDM_OK 1
# endif
#endif
#ifndef TDM_OK
# define TDM_OK 0
#endif

union FragU { v16bf v; uint4 q[2]; };

__device__ __forceinline__ u16 f2bf_rn(float f) {
  unsigned int x = __float_as_uint(f);
  unsigned int r = x + 0x7FFFu + ((x >> 16) & 1u);   // round-to-nearest-even
  return (u16)(r >> 16);
}
__device__ __forceinline__ float bf2f(u16 h) {
  return __uint_as_float(((unsigned int)h) << 16);
}

// ---- bf16 A fragment 16x32 (MxK), bf16 row-major source (K contiguous) ----
__device__ __forceinline__ v16bf load_frag_a(const u16* __restrict__ base,
                                             int ld, int lane) {
  const u16* p = base + (size_t)(lane & 15) * ld + ((lane >> 4) ? 8 : 0);
  FragU f;
  f.q[0] = *(const uint4*)p;
  f.q[1] = *(const uint4*)(p + 16);
  return f.v;
}
// ---- bf16 B fragment 32x16 (KxN): per-lane K = 8h..8h+7, 16+8h..16+8h+7 ---
__device__ __forceinline__ v16bf load_frag_b(const u16* __restrict__ base,
                                             int ld, int lane) {
  const u16* p = base + (size_t)(lane & 15) * ld + ((lane >> 4) ? 8 : 0);
  FragU f;
  f.q[0] = *(const uint4*)p;
  f.q[1] = *(const uint4*)(p + 16);
  return f.v;
}

// bf16x3: (ah+al)(bh+bl) ~= ah*bh + al*bh + ah*bl
__device__ __forceinline__ v8f mma3(v16bf ah, v16bf al, v16bf bh, v16bf bl, v8f acc) {
  acc = __builtin_amdgcn_wmma_f32_16x16x32_bf16(false, ah, false, bh, (short)0, acc, false, false);
  acc = __builtin_amdgcn_wmma_f32_16x16x32_bf16(false, al, false, bh, (short)0, acc, false, false);
  acc = __builtin_amdgcn_wmma_f32_16x16x32_bf16(false, ah, false, bl, (short)0, acc, false, false);
  return acc;
}

// --------------------------- fp32 -> bf16 hi/lo ----------------------------
__global__ __launch_bounds__(256) void k_split(const float* __restrict__ src,
                                               u16* __restrict__ hi,
                                               u16* __restrict__ lo, int n4) {
  int i = blockIdx.x * 256 + threadIdx.x;
  if (i >= n4) return;
  float4 f = ((const float4*)src)[i];
  u16 h0 = f2bf_rn(f.x), h1 = f2bf_rn(f.y), h2 = f2bf_rn(f.z), h3 = f2bf_rn(f.w);
  u16 l0 = f2bf_rn(f.x - bf2f(h0)), l1 = f2bf_rn(f.y - bf2f(h1));
  u16 l2 = f2bf_rn(f.z - bf2f(h2)), l3 = f2bf_rn(f.w - bf2f(h3));
  uint2 hp, lp;
  hp.x = (unsigned)h0 | ((unsigned)h1 << 16); hp.y = (unsigned)h2 | ((unsigned)h3 << 16);
  lp.x = (unsigned)l0 | ((unsigned)l1 << 16); lp.y = (unsigned)l2 | ((unsigned)l3 << 16);
  ((uint2*)hi)[i] = hp;
  ((uint2*)lo)[i] = lp;
}

// ------------------------------- in-proj -----------------------------------
// 64x32 per wave (4x2 WMMA tiles); wave tiles: b*(36*64) + ot*64 + lt
__global__ __launch_bounds__(256) void k_inproj(const u16* __restrict__ iwh,
                                                const u16* __restrict__ iwl,
                                                const u16* __restrict__ uh,
                                                const u16* __restrict__ ul,
                                                float* __restrict__ x) {
  int wave = threadIdx.x >> 5, lane = threadIdx.x & 31;
  int t = blockIdx.x * 8 + wave;                 // 0..9215
  int b  = t / (36 * 64);
  int r  = t % (36 * 64);
  int ot = r / 64, lt = r % 64;
  int o0 = ot * 64, l0 = lt * 32;

  size_t arow[4], brow[2];
#pragma unroll
  for (int i = 0; i < 4; ++i) arow[i] = (size_t)(o0 + 16 * i) * D_;
#pragma unroll
  for (int j = 0; j < 2; ++j) brow[j] = ((size_t)b * L_ + l0 + 16 * j) * D_;

  v8f zero = {0,0,0,0,0,0,0,0};
  v8f acc[4][2];
#pragma unroll
  for (int i = 0; i < 4; ++i)
#pragma unroll
    for (int j = 0; j < 2; ++j) acc[i][j] = zero;

  for (int kc = 0; kc < D_; kc += 32) {
    v16bf ah[4], al[4], bh[2], bl[2];
#pragma unroll
    for (int j = 0; j < 2; ++j) {
      bh[j] = load_frag_b(uh + brow[j] + kc, D_, lane);
      bl[j] = load_frag_b(ul + brow[j] + kc, D_, lane);
    }
#pragma unroll
    for (int i = 0; i < 4; ++i) {
      ah[i] = load_frag_a(iwh + arow[i] + kc, D_, lane);
      al[i] = load_frag_a(iwl + arow[i] + kc, D_, lane);
    }
#pragma unroll
    for (int i = 0; i < 4; ++i)
#pragma unroll
      for (int j = 0; j < 2; ++j)
        acc[i][j] = mma3(ah[i], al[i], bh[j], bl[j], acc[i][j]);
  }

  int n = lane & 15, hb = lane >> 4;
#pragma unroll
  for (int i = 0; i < 4; ++i) {
#pragma unroll
    for (int rr = 0; rr < 8; ++rr) {
      int row = o0 + 16 * i + rr + hb * 8;
      size_t p = ((size_t)b * D3_ + row) * L_ + l0 + n;
      x[p]      = acc[i][0][rr];
      x[p + 16] = acc[i][1][rr];
    }
  }
}

// ------------------------------ short conv ---------------------------------
__global__ __launch_bounds__(256) void k_shortconv(const float* __restrict__ x,
                                                   const float* __restrict__ sw,
                                                   const float* __restrict__ sb,
                                                   float* __restrict__ x1v,
                                                   float* __restrict__ x2o) {
  size_t idx = (size_t)blockIdx.x * 256 + threadIdx.x;  // over B*D*L
  int l = (int)(idx % L_);
  int d = (int)((idx / L_) % D_);
  int b = (int)(idx / ((size_t)L_ * D_));
  float c[3];
#pragma unroll
  for (int g = 0; g < 3; ++g) {
    int ch = g * D_ + d;
    const float* row = x + ((size_t)b * D3_ + ch) * L_;
    float xm = (l > 0)      ? row[l - 1] : 0.f;
    float xc = row[l];
    float xp = (l < L_ - 1) ? row[l + 1] : 0.f;
    c[g] = sw[ch * 3 + 0] * xm + sw[ch * 3 + 1] * xc + sw[ch * 3 + 2] * xp + sb[ch];
  }
  size_t o = ((size_t)b * D_ + d) * L_ + l;
  x1v[o] = c[0] * c[2];
  x2o[o] = c[1];
}

// --------------------------- causal long conv ------------------------------
#if TDM_OK
// Minimal 2-D TDM descriptor: one row of L_ fp32, tile == tensor.
// Group0: count=1 | lds_addr | global_addr[56:0] | type=2 (image)
// Group1: data_size=4B; tensor_dim0=tile_dim0=L_; tensor_dim1=tile_dim1=1;
//         tensor_dim0_stride=L_; everything else 0.
__device__ __forceinline__ void tdm_load_row(const float* gsrc, unsigned lds_off) {
  unsigned long long ga = (unsigned long long)gsrc;
  v4u g0;
  g0[0] = 1u;                                            // count=1 (user desc)
  g0[1] = lds_off;                                       // lds_addr (bytes)
  g0[2] = (unsigned)(ga & 0xFFFFFFFFu);                  // global_addr[31:0]
  g0[3] = (unsigned)((ga >> 32) & 0x01FFFFFFu) | 0x80000000u; // [56:32]|type=2
  v8i g1;
  g1[0] = (int)(2u << 16);                               // data_size=4B
  g1[1] = (int)(((unsigned)L_ & 0xFFFFu) << 16);         // tensor_dim0 lo16
  g1[2] = (int)(((unsigned)L_ >> 16) | (1u << 16));      // dim0 hi16 | dim1 lo16=1
  g1[3] = (int)(((unsigned)L_ & 0xFFFFu) << 16);         // tile_dim0 = L_
  g1[4] = 1;                                             // tile_dim1 = 1
  g1[5] = (int)L_;                                       // dim0_stride lo32
  g1[6] = 0;
  g1[7] = 0;
  v4i z4 = {0, 0, 0, 0};
  v8i z8 = {0, 0, 0, 0, 0, 0, 0, 0};
  __builtin_amdgcn_tensor_load_to_lds(g0, g1, z4, z4, z8, 0);
}
#endif

// Y[:, i] += G_s @ X[:, i-s], G_s[r,c] = filt[16s + r - c] (0 if < 0).
template <bool GA, bool GB>
__device__ __forceinline__ v8f tstep(const float* fs, const float* xs,
                                     int s, int i0, int n, int hb, v8f acc) {
  int fbase = 16 * s;
  int j = i0 + n - s;
  bool jv = !GB || (j >= 0);
  int xb = (GB && !jv) ? 0 : j * 16;
#pragma unroll
  for (int c0 = 0; c0 < 16; c0 += 4) {
    int ka = c0 + 2 * hb;
    int ia0 = fbase + n - ka;
    v2f a, bb;
    if (GA) {
      a.x = (ia0 >= 0)     ? fs[ia0]     : 0.f;
      a.y = (ia0 - 1 >= 0) ? fs[ia0 - 1] : 0.f;
    } else {
      a.x = fs[ia0];
      a.y = fs[ia0 - 1];
    }
    float2 tv = *(const float2*)(xs + xb + ka);     // 8B aligned
    bb.x = (!GB || jv) ? tv.x : 0.f;
    bb.y = (!GB || jv) ? tv.y : 0.f;
    acc = __builtin_amdgcn_wmma_f32_16x16x4_f32(false, a, false, bb, (short)0, acc, false, false);
  }
  return acc;
}

__global__ __launch_bounds__(256) void k_longconv(const float* __restrict__ x1v,
                                                  const float* __restrict__ x2,
                                                  const float* __restrict__ filt,
                                                  u16* __restrict__ y2h,
                                                  u16* __restrict__ y2l) {
  __shared__ float ldsbuf[2 * L_];
  float* xs = ldsbuf;            // LDS offset 0
  float* fs = ldsbuf + L_;       // LDS offset L_*4 bytes
  int bd = blockIdx.x;           // 0..B*D-1
  int b = bd / D_, d = bd % D_;

#if TDM_OK
  if (threadIdx.x < 32) {        // one wave issues the two row DMAs
    tdm_load_row(x1v + (size_t)bd * L_, 0u);
    tdm_load_row(filt + (size_t)d * L_, (unsigned)(L_ * sizeof(float)));
    __builtin_amdgcn_s_wait_tensorcnt(0);
  }
#else
  {
    const float4* xg = (const float4*)(x1v + (size_t)bd * L_);
    const float4* fg = (const float4*)(filt + (size_t)d * L_);
    for (int i = threadIdx.x; i < L_ / 4; i += 256) {
      ((float4*)xs)[i] = xg[i];
      ((float4*)fs)[i] = fg[i];
    }
  }
#endif
  __syncthreads();
  // The TDM engine wrote ldsbuf through a descriptor the compiler cannot see.
  // ldsbuf is an internal, otherwise store-free LDS global, so without an
  // address ESCAPE LLVM folds all its loads to undef (observed ds=0).
  // Pass the address into an asm with a memory clobber: now the asm may have
  // stored through it, so subsequent ds_loads must be materialized. Keep
  // reading via the original pointers so addrspace(3) inference survives.
  {
    float* esc = ldsbuf;
    asm volatile("" : "+v"(esc) :: "memory");
  }

  int wave = threadIdx.x >> 5, lane = threadIdx.x & 31;
  int i0 = wave * 16;            // first block-column of this wave's tile
  int n = lane & 15, hb = lane >> 4;

  v8f acc = {0,0,0,0,0,0,0,0};
  acc = tstep<true, false>(fs, xs, 0, i0, n, hb, acc);      // s = 0
  for (int s = 1; s <= i0; ++s)                             // guard-free bulk
    acc = tstep<false, false>(fs, xs, s, i0, n, hb, acc);
  for (int s = i0 + 1; s <= i0 + 15; ++s)                   // causal tail
    acc = tstep<false, true>(fs, xs, s, i0, n, hb, acc);

  // gate with x2, write bf16 hi/lo transposed to (B, L, D)
  const float* x2row = x2 + (size_t)bd * L_;
  size_t obase = (size_t)b * L_ * D_ + d;
#pragma unroll
  for (int rr = 0; rr < 8; ++rr) {
    int l = (i0 + n) * 16 + rr + 8 * hb;
    float val = acc[rr] * x2row[l];
    u16 h = f2bf_rn(val);
    y2h[obase + (size_t)l * D_] = h;
    y2l[obase + (size_t)l * D_] = f2bf_rn(val - bf2f(h));
  }
}

// ------------------------------- out-proj ----------------------------------
// 64x32 per wave; wave tiles: b*(32*24) + lt*24 + ot
__global__ __launch_bounds__(256) void k_outproj(const u16* __restrict__ y2h,
                                                 const u16* __restrict__ y2l,
                                                 const u16* __restrict__ owh,
                                                 const u16* __restrict__ owl,
                                                 const float* __restrict__ out_b,
                                                 float* __restrict__ out) {
  int wave = threadIdx.x >> 5, lane = threadIdx.x & 31;
  int t = blockIdx.x * 8 + wave;                 // 0..3071
  int b  = t / (32 * 24);
  int r  = t % (32 * 24);
  int lt = r / 24, ot = r % 24;
  int l0 = lt * 64, o0 = ot * 32;

  size_t arow[4], brow[2];
#pragma unroll
  for (int i = 0; i < 4; ++i) arow[i] = ((size_t)b * L_ + l0 + 16 * i) * D_;
#pragma unroll
  for (int j = 0; j < 2; ++j) brow[j] = (size_t)(o0 + 16 * j) * D_;

  v8f zero = {0,0,0,0,0,0,0,0};
  v8f acc[4][2];
#pragma unroll
  for (int i = 0; i < 4; ++i)
#pragma unroll
    for (int j = 0; j < 2; ++j) acc[i][j] = zero;

  for (int kc = 0; kc < D_; kc += 32) {
    v16bf ah[4], al[4], bh[2], bl[2];
#pragma unroll
    for (int j = 0; j < 2; ++j) {
      bh[j] = load_frag_b(owh + brow[j] + kc, D_, lane);
      bl[j] = load_frag_b(owl + brow[j] + kc, D_, lane);
    }
#pragma unroll
    for (int i = 0; i < 4; ++i) {
      ah[i] = load_frag_a(y2h + arow[i] + kc, D_, lane);
      al[i] = load_frag_a(y2l + arow[i] + kc, D_, lane);
    }
#pragma unroll
    for (int i = 0; i < 4; ++i)
#pragma unroll
      for (int j = 0; j < 2; ++j)
        acc[i][j] = mma3(ah[i], al[i], bh[j], bl[j], acc[i][j]);
  }

  int n = lane & 15, hb = lane >> 4;
  float bias0 = out_b[o0 + n];
  float bias1 = out_b[o0 + 16 + n];
#pragma unroll
  for (int i = 0; i < 4; ++i) {
#pragma unroll
    for (int rr = 0; rr < 8; ++rr) {
      int l = l0 + 16 * i + rr + hb * 8;
      size_t p = ((size_t)b * L_ + l) * D_ + o0 + n;
      out[p]      = acc[i][0][rr] + bias0;
      out[p + 16] = acc[i][1][rr] + bias1;
    }
  }
}

// ------------------------------- launcher ----------------------------------
extern "C" void kernel_launch(void* const* d_in, const int* in_sizes, int n_in,
                              void* d_out, int out_size, void* d_ws, size_t ws_size,
                              hipStream_t stream) {
  const float* u       = (const float*)d_in[0];
  const float* in_w    = (const float*)d_in[1];
  const float* short_w = (const float*)d_in[2];
  const float* short_b = (const float*)d_in[3];
  const float* filt    = (const float*)d_in[4];
  const float* out_w   = (const float*)d_in[5];
  const float* out_b   = (const float*)d_in[6];
  float* out = (float*)d_out;

  const size_t NBDL = (size_t)B_ * D_ * L_;        // 6,291,456
  const size_t NX   = (size_t)B_ * D3_ * L_;       // 18,874,368
  const size_t NIW  = (size_t)D3_ * D_;            // 1,769,472
  const size_t NOW  = (size_t)D_ * D_;             // 589,824

  float* x   = (float*)d_ws;                       // (B, 3D, L)
  float* x1v = x + NX;                             // (B, D, L)
  float* x2  = x1v + NBDL;                         // (B, D, L)
  u16* uh  = (u16*)(x2 + NBDL);                    // (B, L, D) bf16 hi
  u16* ul  = uh + NBDL;                            // (B, L, D) bf16 lo
  u16* iwh = ul + NBDL;                            // (3D, D)
  u16* iwl = iwh + NIW;
  u16* owh = iwl + NIW;                            // (D, D)
  u16* owl = owh + NOW;
  u16* y2h = owl + NOW;                            // (B, L, D)
  u16* y2l = y2h + NBDL;

  k_split<<<6144, 256, 0, stream>>>(u, uh, ul, (int)(NBDL / 4));
  k_split<<<1728, 256, 0, stream>>>(in_w, iwh, iwl, (int)(NIW / 4));
  k_split<<<576,  256, 0, stream>>>(out_w, owh, owl, (int)(NOW / 4));

  k_inproj   <<<1152,  256, 0, stream>>>(iwh, iwl, uh, ul, x);
  k_shortconv<<<24576, 256, 0, stream>>>(x, short_w, short_b, x1v, x2);
  k_longconv <<<3072,  256, 0, stream>>>(x1v, x2, filt, y2h, y2l);
  k_outproj  <<<384,   256, 0, stream>>>(y2h, y2l, owh, owl, out_b, out);
}
